// Dense_test_2layer_30940944400812
// MI455X (gfx1250) — matrix operand order repaired and checked by reading, not compiled, not run on hardware
//
#include <hip/hip_runtime.h>
#include <math.h>

typedef __bf16 bf16_t;
typedef __attribute__((ext_vector_type(16))) __bf16 v16bf;
typedef __attribute__((ext_vector_type(8)))  __bf16 v8bf;
typedef __attribute__((ext_vector_type(8)))  float  v8f;

#define VTH      1.0f
#define B_TOT    256
#define T_STEPS  500
#define DIN      700
#define NN       64
#define NBR      8
#define NOUT     20
#define IN1      768          // 700 + 64 + 4 pad
#define IN2      128          // 64 + 64
#define NCOL     512          // N * BR
#define NKK1     (IN1/32)     // 24 K-steps for GEMM1
#define NKK2     (IN2/32)     // 4  K-steps for GEMM2
#define BT       16           // batch rows per workgroup
#define NTHREADS 512          // 16 waves (wave32)

__device__ __forceinline__ float sigmoidf_(float x) { return 1.0f / (1.0f + __expf(-x)); }

union ABFrag { v16bf v; v8bf h[2]; };
union Pack4 { bf16_t b[4]; uint2 u; };

// ---------------------------------------------------------------------------
// Repack f32 row-major weights [NCOL x K] into bf16 WMMA B-fragment order.
// Flat fragment id = (tile*nkk + kk)*32 + lane; each lane owns 16 contiguous
// K values of one output column (code's own k order, matching A: N striped across lanes, lanes 0-15
// hold K[0..15] of the 32-K slab, lanes 16-31 hold K[16..31]).
// ---------------------------------------------------------------------------
__global__ void repack_kernel(const float* __restrict__ W, bf16_t* __restrict__ Wp,
                              int nkk, int K) {
    int gid = blockIdx.x * blockDim.x + threadIdx.x;
    int total = 32 * nkk * 32;
    if (gid >= total) return;
    int lane = gid & 31;
    int grp  = gid >> 5;
    int kk   = grp % nkk;
    int tile = grp / nkk;
    int c  = tile * 16 + (lane & 15);
    int kb = kk * 32 + (lane >> 4) * 16;
    const float* src = W + (size_t)c * K + kb;
    bf16_t* dst = Wp + (size_t)gid * 16;
#pragma unroll
    for (int i = 0; i < 16; ++i) dst[i] = (bf16_t)src[i];
}

// ---------------------------------------------------------------------------
// Persistent SNN kernel: one workgroup owns BT=16 batch rows for all T steps.
// 16 waves each own 2 WMMA N-tiles (= 4 neurons x 8 branches) of both GEMMs.
// Dendritic state d lives in WMMA C-layout registers; spikes round-trip LDS.
// W1 B-fragments are streamed from L2 each step through an opaque (anti-LICM)
// scalar offset; W2 B-fragments (64 VGPRs) are register-resident for all steps.
// ---------------------------------------------------------------------------
__global__ __launch_bounds__(NTHREADS)
void snn_kernel(const float* __restrict__ x,
                const float* __restrict__ b1, const float* __restrict__ tau_m1,
                const float* __restrict__ tau_n1,
                const float* __restrict__ b2, const float* __restrict__ tau_m2,
                const float* __restrict__ tau_n2,
                const float* __restrict__ W3, const float* __restrict__ b3,
                const float* __restrict__ tau_m3,
                const float* __restrict__ mem1_0, const float* __restrict__ mem2_0,
                const bf16_t* __restrict__ W1p, const bf16_t* __restrict__ W2p,
                float* __restrict__ out)
{
    __shared__ bf16_t lds_k1[BT * IN1];   // [x_t | sp1 | pad] activations (A of GEMM1)
    __shared__ bf16_t lds_k2[BT * IN2];   // [sp1 | sp2]               (A of GEMM2)
    __shared__ float  lds_w3[NOUT * NN];
    __shared__ float  lds_m3[BT * NOUT];

    const int tid     = threadIdx.x;
    const int lane    = tid & 31;
    const int wave    = tid >> 5;
    const int r0      = blockIdx.x * BT;
    const int t0      = 2 * wave;            // first of this wave's two N-tiles
    const int cofs    = lane & 15;           // column-within-tile == A-matrix row
    const int rowbase = (lane >> 4) * 8;     // C-layout: lanes>=16 hold rows 8..15

    // ---- init LDS ----
    for (int i = tid; i < NOUT * NN; i += NTHREADS) lds_w3[i] = W3[i];
    for (int i = tid; i < BT * IN2;  i += NTHREADS) lds_k2[i] = (bf16_t)0.0f;
    for (int i = tid; i < BT * (IN1 - DIN); i += NTHREADS) {
        int row = i / (IN1 - DIN);
        int k   = i - row * (IN1 - DIN);
        lds_k1[row * IN1 + DIN + k] = (bf16_t)0.0f;   // sp1 slot + pad = 0
    }

    // ---- per-lane constants & persistent state (WMMA C layout) ----
    float al1[2], be1[2], bi1[2], al2[2], be2[2], bi2[2];
    float d1s[2][8], d2s[2][8], m1[2][8], m2[2][8], s1[2][8], s2[2][8];
#pragma unroll
    for (int ti = 0; ti < 2; ++ti) {
        int c = 16 * (t0 + ti) + cofs;       // flat column in [0,512): c = n*8 + j
        int n = c >> 3;                      // neuron
        be1[ti] = sigmoidf_(tau_n1[c]);      // tau_n is [N,BR] flat == c
        be2[ti] = sigmoidf_(tau_n2[c]);
        bi1[ti] = b1[c];
        bi2[ti] = b2[c];
        al1[ti] = sigmoidf_(tau_m1[n]);
        al2[ti] = sigmoidf_(tau_m2[n]);
#pragma unroll
        for (int v = 0; v < 8; ++v) {
            m1[ti][v] = mem1_0[(size_t)(r0 + rowbase + v) * NN + n];
            m2[ti][v] = mem2_0[(size_t)(r0 + rowbase + v) * NN + n];
            d1s[ti][v] = 0.f; d2s[ti][v] = 0.f;
            s1[ti][v]  = 0.f; s2[ti][v]  = 0.f;
        }
    }

    // ---- W2 B-fragments: small enough (64 VGPRs) to pin in registers ----
    v16bf w2frag[2][NKK2];
#pragma unroll
    for (int ti = 0; ti < 2; ++ti)
#pragma unroll
        for (int kk = 0; kk < NKK2; ++kk)
            w2frag[ti][kk] = *(const v16bf*)(W2p +
                ((size_t)(((t0 + ti) * NKK2 + kk) * 32 + lane)) * 16);

    // ---- readout owner threads (one per (row, out)) ----
    const bool owner = (tid < BT * NOUT);
    const int  orow  = tid / NOUT;
    const int  oo    = tid - orow * NOUT;
    float m3 = 0.f, accr = 0.f, al3 = 0.f, bi3 = 0.f;
    if (owner) { al3 = sigmoidf_(tau_m3[oo]); bi3 = b3[oo]; }

    __syncthreads();

    const bf16_t* k1row = lds_k1 + cofs * IN1;
    const bf16_t* k2row = lds_k2 + cofs * IN2;
    const int asel = (lane >> 4) * 16;        // K sub-slab select for A frags

    for (int t = 0; t < T_STEPS; ++t) {
        // ---- phase a: stage x_t (f32 -> bf16) into lds_k1[:,0:700] ----
        for (int e = tid; e < BT * (DIN / 4); e += NTHREADS) {
            int row = e / (DIN / 4);
            int k4  = e - row * (DIN / 4);
            const float4 val =
                *(const float4*)(x + ((size_t)(r0 + row) * T_STEPS + t) * DIN + k4 * 4);
            Pack4 p;
            p.b[0] = (bf16_t)val.x; p.b[1] = (bf16_t)val.y;
            p.b[2] = (bf16_t)val.z; p.b[3] = (bf16_t)val.w;
            *(uint2*)(lds_k1 + row * IN1 + k4 * 4) = p.u;   // one ds_store_b64
        }
        __syncthreads(); // A

        // Opaque scalar offset (always 0) re-defined each iteration: defeats
        // LICM so B-fragments stream from L2 every step (no giant register
        // cache + scratch spill), while keeping GEP provenance on W1p so the
        // loads lower to global_load_b128 (LOADcnt only), not flat_load.
        int zoff = 0;
        asm volatile("" : "+s"(zoff));
        const bf16_t* w1p_t = W1p + zoff;

        // ---- phase b: GEMM1  pre1 = k1 @ W1.T  (bf16 WMMA, f32 acc) ----
        v8f acc[2];
#pragma unroll
        for (int ti = 0; ti < 2; ++ti)
#pragma unroll
            for (int v = 0; v < 8; ++v) acc[ti][v] = 0.f;

        for (int kk = 0; kk < NKK1; ++kk) {
            int kb = kk * 32;
            ABFrag a;
            a.h[0] = *(const v8bf*)(k1row + kb + asel);
            a.h[1] = *(const v8bf*)(k1row + kb + 8 + asel);
#pragma unroll
            for (int ti = 0; ti < 2; ++ti) {
                v16bf bfrag = *(const v16bf*)(w1p_t +
                    ((size_t)(((t0 + ti) * NKK1 + kk) * 32 + lane)) * 16);
                acc[ti] = __builtin_amdgcn_wmma_f32_16x16x32_bf16(
                    false, a.v, false, bfrag, (short)0, acc[ti], false, false);
            }
        }
        __syncthreads(); // B: all GEMM1 reads of lds_k1 done before sp1 writes

        // ---- phase c: dendrite/LIF update layer1, emit spikes ----
#pragma unroll
        for (int ti = 0; ti < 2; ++ti) {
#pragma unroll
            for (int v = 0; v < 8; ++v) {
                float pre = acc[ti][v] + bi1[ti];
                float d = be1[ti] * d1s[ti][v] + (1.f - be1[ti]) * pre;
                d1s[ti][v] = d;
                float l = d;                         // branch sum over 8 lanes
                l += __shfl_xor(l, 1);
                l += __shfl_xor(l, 2);
                l += __shfl_xor(l, 4);
                float m = m1[ti][v] * al1[ti] + (1.f - al1[ti]) * l - VTH * s1[ti][v];
                m1[ti][v] = m;
                s1[ti][v] = (m - VTH) > 0.f ? 1.f : 0.f;
            }
            if ((lane & 7) == 0) {                   // one lane per neuron writes
                int n = 2 * (t0 + ti) + (cofs >> 3);
#pragma unroll
                for (int v = 0; v < 8; ++v) {
                    bf16_t sb = (bf16_t)s1[ti][v];
                    lds_k1[(rowbase + v) * IN1 + DIN + n] = sb; // for k1 @ t+1
                    lds_k2[(rowbase + v) * IN2 + n]       = sb; // for GEMM2 now
                }
            }
        }
        __syncthreads(); // C

        // ---- phase d: GEMM2  pre2 = [sp1|sp2] @ W2.T (register-resident B) ----
#pragma unroll
        for (int ti = 0; ti < 2; ++ti)
#pragma unroll
            for (int v = 0; v < 8; ++v) acc[ti][v] = 0.f;

#pragma unroll
        for (int kk = 0; kk < NKK2; ++kk) {
            int kb = kk * 32;
            ABFrag a;
            a.h[0] = *(const v8bf*)(k2row + kb + asel);
            a.h[1] = *(const v8bf*)(k2row + kb + 8 + asel);
#pragma unroll
            for (int ti = 0; ti < 2; ++ti) {
                acc[ti] = __builtin_amdgcn_wmma_f32_16x16x32_bf16(
                    false, a.v, false, w2frag[ti][kk], (short)0, acc[ti], false, false);
            }
        }
        __syncthreads(); // D

        // ---- phase e: dendrite/LIF update layer2, emit sp2 ----
#pragma unroll
        for (int ti = 0; ti < 2; ++ti) {
#pragma unroll
            for (int v = 0; v < 8; ++v) {
                float pre = acc[ti][v] + bi2[ti];
                float d = be2[ti] * d2s[ti][v] + (1.f - be2[ti]) * pre;
                d2s[ti][v] = d;
                float l = d;
                l += __shfl_xor(l, 1);
                l += __shfl_xor(l, 2);
                l += __shfl_xor(l, 4);
                float m = m2[ti][v] * al2[ti] + (1.f - al2[ti]) * l - VTH * s2[ti][v];
                m2[ti][v] = m;
                s2[ti][v] = (m - VTH) > 0.f ? 1.f : 0.f;
            }
            if ((lane & 7) == 0) {
                int n = 2 * (t0 + ti) + (cofs >> 3);
#pragma unroll
                for (int v = 0; v < 8; ++v)
                    lds_k2[(rowbase + v) * IN2 + NN + n] = (bf16_t)s2[ti][v];
            }
        }
        __syncthreads(); // E

        // ---- phase f: leaky readout  mem3 = a3*mem3 + (1-a3)*(sp2 @ W3.T + b3) ----
        if (owner) {
            float dot = bi3;
            const v8bf*   sp2v = (const v8bf*)(lds_k2 + orow * IN2 + NN);
            const float4* w3v  = (const float4*)(lds_w3 + oo * NN);
#pragma unroll
            for (int c8 = 0; c8 < 8; ++c8) {        // 64 = 8 x v8bf chunks
                v8bf s = sp2v[c8];
                float4 wa = w3v[c8 * 2 + 0];
                float4 wb = w3v[c8 * 2 + 1];
                dot += (float)s[0] * wa.x + (float)s[1] * wa.y +
                       (float)s[2] * wa.z + (float)s[3] * wa.w +
                       (float)s[4] * wb.x + (float)s[5] * wb.y +
                       (float)s[6] * wb.z + (float)s[7] * wb.w;
            }
            m3 = m3 * al3 + (1.f - al3) * dot;
            lds_m3[orow * NOUT + oo] = m3;
        }
        __syncthreads(); // F

        // ---- phase g: softmax accumulate for t > 10 ----
        if (owner && t > 10) {
            float mx = -1e30f;
#pragma unroll
            for (int j = 0; j < NOUT; ++j) mx = fmaxf(mx, lds_m3[orow * NOUT + j]);
            float se = 0.f;
#pragma unroll
            for (int j = 0; j < NOUT; ++j) se += __expf(lds_m3[orow * NOUT + j] - mx);
            accr += __expf(m3 - mx) / se;
        }

        // prefetch next timestep's x rows into cache
        if (t + 1 < T_STEPS) {
            for (int e = tid; e < BT * (DIN / 4); e += NTHREADS * 8) {
                int row = e / (DIN / 4);
                int k4  = e - row * (DIN / 4);
                __builtin_prefetch(
                    x + ((size_t)(r0 + row) * T_STEPS + (t + 1)) * DIN + k4 * 4, 0, 0);
            }
        }
    }

    if (owner) out[(size_t)(r0 + orow) * NOUT + oo] = accr;
}

// ---------------------------------------------------------------------------
extern "C" void kernel_launch(void* const* d_in, const int* in_sizes, int n_in,
                              void* d_out, int out_size, void* d_ws, size_t ws_size,
                              hipStream_t stream) {
    (void)in_sizes; (void)n_in; (void)out_size; (void)ws_size;
    const float* x      = (const float*)d_in[0];
    const float* W1     = (const float*)d_in[1];
    const float* b1     = (const float*)d_in[2];
    const float* tau_m1 = (const float*)d_in[3];
    const float* tau_n1 = (const float*)d_in[4];
    const float* W2     = (const float*)d_in[5];
    const float* b2     = (const float*)d_in[6];
    const float* tau_m2 = (const float*)d_in[7];
    const float* tau_n2 = (const float*)d_in[8];
    const float* W3     = (const float*)d_in[9];
    const float* b3     = (const float*)d_in[10];
    const float* tau_m3 = (const float*)d_in[11];
    const float* mem1_0 = (const float*)d_in[12];
    const float* mem2_0 = (const float*)d_in[13];
    float* out = (float*)d_out;

    bf16_t* W1p = (bf16_t*)d_ws;                                   // 512*768 bf16 = 768 KB
    bf16_t* W2p = (bf16_t*)((char*)d_ws + (size_t)NCOL * IN1 * 2); // 512*128 bf16

    int tot1 = 32 * NKK1 * 32;
    int tot2 = 32 * NKK2 * 32;
    repack_kernel<<<(tot1 + 255) / 256, 256, 0, stream>>>(W1, W1p, NKK1, IN1);
    repack_kernel<<<(tot2 + 255) / 256, 256, 0, stream>>>(W2, W2p, NKK2, IN2);

    snn_kernel<<<B_TOT / BT, NTHREADS, 0, stream>>>(
        x, b1, tau_m1, tau_n1, b2, tau_m2, tau_n2,
        W3, b3, tau_m3, mem1_0, mem2_0, W1p, W2p, out);
}
